// SparseCNNTokenizer_5463198400784
// MI455X (gfx1250) — matrix operand order repaired and checked by reading, not compiled, not run on hardware
//
#include <hip/hip_runtime.h>

// ---------------------------------------------------------------------------
// SparK-style sparse CNN tokenizer for MI455X (gfx1250, wave32).
// Pipeline: cvt_w2 -> conv1(mask,relu) -> maxblurpool1 -> conv2 (bf16 WMMA
// implicit GEMM, fused mask+relu) -> maxblurpool2 (+ transpose to [B,196,768]).
// conv2 carries ~94% of FLOPs (242 GFLOP) and runs on v_wmma_f32_16x16x32_bf16
// with ping-pong LDS, async global->LDS weight staging (ASYNCcnt), and one
// barrier per K=64 macro-step (8 WMMAs/wave per barrier).
// ---------------------------------------------------------------------------

typedef __attribute__((ext_vector_type(16))) __bf16 v16bf;
typedef __attribute__((ext_vector_type(8)))  float  v8f;

union U16 { v16bf v; uint4 q[2]; };

__device__ __forceinline__ unsigned short f32_to_bf16(float f) {
  union { float f; unsigned u; } v; v.f = f;
  unsigned r = v.u + 0x7FFFu + ((v.u >> 16) & 1u);   // round-to-nearest-even
  return (unsigned short)(r >> 16);
}
__device__ __forceinline__ float bf16_to_f32(unsigned short h) {
  union { unsigned u; float f; } v; v.u = ((unsigned)h) << 16;
  return v.f;
}

// ---------------------------------------------------------------------------
// Kernel 0: w2 [768][64][7][7] f32 -> bf16 (layout IS already the GEMM-B
// [N=768][K=3136] layout with k = c*49 + kr*7 + kc).
// ---------------------------------------------------------------------------
__global__ void cvt_w2(const float* __restrict__ w2,
                       unsigned short* __restrict__ w2b, int n) {
  int i = blockIdx.x * blockDim.x + threadIdx.x;
  if (i < n) w2b[i] = f32_to_bf16(w2[i]);
}

// ---------------------------------------------------------------------------
// Kernel 1: conv1 7x7 s2 p3, 3->64, fused input-mask (patch=16 on 224) and
// output mask (factor 8 on 112) + ReLU.  One 16x16 output tile per block,
// input tile (37x37x3, masked) + 147-tap filter staged in LDS.
// X1: [64][64][112][112] bf16
// ---------------------------------------------------------------------------
__global__ __launch_bounds__(256) void conv1_mask_relu(
    const float* __restrict__ x, const unsigned char* __restrict__ masks,
    const float* __restrict__ w1, unsigned short* __restrict__ X1) {
  __shared__ float sIn[3 * 37 * 38];   // width padded to 38 (bank friendliness)
  __shared__ float sW[148];

  const int tid = threadIdx.y * 16 + threadIdx.x;
  const int bz = blockIdx.z;
  const int b = bz >> 6, oc = bz & 63;
  const int oh0 = blockIdx.y * 16, ow0 = blockIdx.x * 16;
  const int ih0 = 2 * oh0 - 3,     iw0 = 2 * ow0 - 3;

  for (int i = tid; i < 147; i += 256) sW[i] = w1[oc * 147 + i];
  for (int i = tid; i < 3 * 37 * 37; i += 256) {
    int c = i / 1369, rem = i - c * 1369;
    int r = rem / 37, col = rem - r * 37;
    int ih = ih0 + r, iw = iw0 + col;
    float v = 0.f;
    if ((unsigned)ih < 224u && (unsigned)iw < 224u) {
      if (masks[b * 196 + (ih >> 4) * 14 + (iw >> 4)])
        v = x[(((size_t)b * 3 + c) * 224 + ih) * 224 + iw];
    }
    sIn[c * 1406 + r * 38 + col] = v;
  }
  __syncthreads();

  float acc = 0.f;
#pragma unroll
  for (int c = 0; c < 3; ++c)
#pragma unroll
    for (int kr = 0; kr < 7; ++kr)
#pragma unroll
      for (int kc = 0; kc < 7; ++kc)
        acc = fmaf(sIn[c * 1406 + (2 * threadIdx.y + kr) * 38 + 2 * threadIdx.x + kc],
                   sW[c * 49 + kr * 7 + kc], acc);

  const int oh = oh0 + threadIdx.y, ow = ow0 + threadIdx.x;
  const unsigned char mk = masks[b * 196 + (oh >> 3) * 14 + (ow >> 3)];
  const float v = mk ? fmaxf(acc, 0.f) : 0.f;
  X1[(((size_t)b * 64 + oc) * 112 + oh) * 112 + ow] = f32_to_bf16(v);
}

// ---------------------------------------------------------------------------
// Kernel 2: MaxBlurPool 112 -> 56.  maxpool(k2,s1) + depthwise binomial blur
// (k3,s2,p1) reads exactly a 4x4 window per output; kernel is separable
// [1,2,1]/4 x [1,2,1]/4.  X1p: [64][64][56][56] bf16.
// ---------------------------------------------------------------------------
__global__ void maxblurpool1(const unsigned short* __restrict__ X1,
                             unsigned short* __restrict__ X1p) {
  int i = blockIdx.x * blockDim.x + threadIdx.x;
  if (i >= 64 * 64 * 56 * 56) return;
  const int w = i % 56, h = (i / 56) % 56;
  const int bc = i / 3136;
  const unsigned short* src = X1 + (size_t)bc * 112 * 112;
  const float kb[3] = {0.25f, 0.5f, 0.25f};
  float acc = 0.f;
#pragma unroll
  for (int t = 0; t < 3; ++t) {
    const int r = 2 * h - 1 + t;
    if ((unsigned)r > 110u) continue;      // zero-padded blur input
#pragma unroll
    for (int u = 0; u < 3; ++u) {
      const int s = 2 * w - 1 + u;
      if ((unsigned)s > 110u) continue;
      float m00 = bf16_to_f32(src[r * 112 + s]);
      float m01 = bf16_to_f32(src[r * 112 + s + 1]);
      float m10 = bf16_to_f32(src[(r + 1) * 112 + s]);
      float m11 = bf16_to_f32(src[(r + 1) * 112 + s + 1]);
      acc += kb[t] * kb[u] * fmaxf(fmaxf(m00, m01), fmaxf(m10, m11));
    }
  }
  X1p[i] = f32_to_bf16(acc);
}

// ---------------------------------------------------------------------------
// Kernel 3: conv2 7x7 s2 p3, 64->768 as implicit GEMM on bf16 WMMA.
// Per batch: M=784 pixels, N=768 channels, K=3136.  WG = 8 waves, tile
// 64(M)x128(N); each wave owns 32x32 (2x2 fragments).  K macro-step = 64
// (2 WMMA sub-steps -> 8 v_wmma per wave per barrier), 49 steps.
// Ping-pong LDS buffers: B tile streamed with GLOBAL_LOAD_ASYNC_TO_LDS_B128
// (ASYNCcnt, no VGPR round-trip), A tile register-staged im2col (address
// math co-executes under the WMMAs).  One s_barrier per macro-step.
// X2: [64][768][28][28] bf16.
// ---------------------------------------------------------------------------
#define TILE_M 64
#define TILE_N 128
#define TILE_K 64
#define KTOT   3136
#define NSTEPS (KTOT / TILE_K)   // 49

__global__ __launch_bounds__(256) void conv2_wmma(
    const unsigned short* __restrict__ X1p,   // [64][64][56][56] bf16
    const unsigned short* __restrict__ w2b,   // [768][3136] bf16
    const unsigned char* __restrict__ masks,  // [64][196]
    unsigned short* __restrict__ X2) {        // [64][768][28][28] bf16
  __shared__ unsigned short lA[2][TILE_M * TILE_K];   // [m][k]  2 x 8KB
  __shared__ unsigned short lB[2][TILE_N * TILE_K];   // [n][k]  2 x 16KB

  const int tid    = threadIdx.x;
  const int lane   = tid & 31;
  const int wave   = tid >> 5;
  const int lanelo = lane & 15;
  const int hi     = lane >> 4;

  const int b  = blockIdx.z;
  const int m0 = blockIdx.x * TILE_M;
  const int n0 = blockIdx.y * TILE_N;
  const int waveM = wave & 1;    // 2 waves along M (32 rows each)
  const int waveN = wave >> 1;   // 4 waves along N (32 cols each)

  // A staging: each thread produces 16 consecutive k for one m (im2col).
  const int a_m = tid >> 2;            // 0..63
  const int a_k = (tid & 3) * 16;      // 0,16,32,48
  const int am  = m0 + a_m;
  const int aoh = am / 28, aow = am % 28;

  // B staging: each thread owns 32 consecutive k (64B) of one n row.
  const int b_n = tid >> 1;            // 0..127
  const int b_k = (tid & 1) * 32;      // 0 or 32
  const unsigned short* bsrc = &w2b[(size_t)(n0 + b_n) * KTOT + b_k];
  const unsigned ldsB[2] = {
      (unsigned)(uintptr_t)&lB[0][b_n * TILE_K + b_k],
      (unsigned)(uintptr_t)&lB[1][b_n * TILE_K + b_k]};

  const v8f vzero = {0.f, 0.f, 0.f, 0.f, 0.f, 0.f, 0.f, 0.f};
  v8f acc[2][2];
#pragma unroll
  for (int i = 0; i < 2; ++i)
#pragma unroll
    for (int j = 0; j < 2; ++j) acc[i][j] = vzero;

  alignas(16) unsigned short ra[16];

  auto loadA = [&](int k0) {
#pragma unroll
    for (int j = 0; j < 16; ++j) {
      unsigned short v = 0;
      if (am < 784) {
        int kk = k0 + a_k + j;
        int c  = kk / 49;
        int rr = kk - c * 49;
        int kr = rr / 7;
        int kc = rr - kr * 7;
        int ih = 2 * aoh - 3 + kr;
        int iw = 2 * aow - 3 + kc;
        if ((unsigned)ih < 56u && (unsigned)iw < 56u)
          v = X1p[(((size_t)b * 64 + c) * 56 + ih) * 56 + iw];
      }
      ra[j] = v;
    }
  };
  auto storeA = [&](int p) {
    uint4* dst = (uint4*)&lA[p][a_m * TILE_K + a_k];
    dst[0] = ((const uint4*)ra)[0];
    dst[1] = ((const uint4*)ra)[1];
  };
  // Async global->LDS: 64B per thread as 4x B128, shared addr pair, LDS and
  // global both advanced by the instruction offset (ISA 15.18.3).
  auto asyncB = [&](int p, int k0) {
    unsigned la = ldsB[p];
    unsigned long long ga = (unsigned long long)(uintptr_t)(bsrc + k0);
    asm volatile(
        "global_load_async_to_lds_b128 %0, %1, off\n\t"
        "global_load_async_to_lds_b128 %0, %1, off offset:16\n\t"
        "global_load_async_to_lds_b128 %0, %1, off offset:32\n\t"
        "global_load_async_to_lds_b128 %0, %1, off offset:48"
        :: "v"(la), "v"(ga) : "memory");
  };

  // Prologue: tile 0 into buffer 0.
  asyncB(0, 0);
  loadA(0);
  storeA(0);
  asm volatile("s_wait_asynccnt 0x0" ::: "memory");
  __syncthreads();

  for (int ks = 0; ks < NSTEPS; ++ks) {
    const int p = ks & 1;
    const bool hasNext = (ks + 1) < NSTEPS;
    if (hasNext) {
      asyncB(p ^ 1, (ks + 1) * TILE_K);   // DMA weights straight into LDS
      loadA((ks + 1) * TILE_K);           // global gathers in flight under WMMA
    }

#pragma unroll
    for (int t = 0; t < 2; ++t) {         // two K=32 WMMA sub-steps
      U16 af[2], bfr[2];
      // A frag: lane<16 holds k {0..15}, lane>=16 holds {16..31}
#pragma unroll
      for (int i = 0; i < 2; ++i) {
        const unsigned short* ap =
            &lA[p][(waveM * 32 + i * 16 + lanelo) * TILE_K + t * 32];
        af[i].q[0] = *(const uint4*)(ap + 16 * hi);
        af[i].q[1] = *(const uint4*)(ap + 16 * hi + 8);
      }
      // B frag: lane holds column n=lanelo, k = 16*hi .. 16*hi+15
#pragma unroll
      for (int j = 0; j < 2; ++j) {
        const unsigned short* bp =
            &lB[p][(waveN * 32 + j * 16 + lanelo) * TILE_K + t * 32 + 16 * hi];
        bfr[j].q[0] = *(const uint4*)(bp);
        bfr[j].q[1] = *(const uint4*)(bp + 8);
      }
#pragma unroll
      for (int i = 0; i < 2; ++i)
#pragma unroll
        for (int j = 0; j < 2; ++j)
          acc[i][j] = __builtin_amdgcn_wmma_f32_16x16x32_bf16(
              false, af[i].v, false, bfr[j].v, (short)0, acc[i][j], false, false);
    }

    if (hasNext) {
      storeA(p ^ 1);                                  // regs -> next buffer
      asm volatile("s_wait_asynccnt 0x0" ::: "memory");  // B DMA landed
    }
    __syncthreads();                                  // one barrier per step
  }

  // Epilogue: D layout — VGPR r: lanes 0-15 M=r, lanes 16-31 M=8+r; N=lane&15.
  const int mbase = m0 + waveM * 32;
  const int nbase = n0 + waveN * 32;
#pragma unroll
  for (int i = 0; i < 2; ++i) {
#pragma unroll
    for (int j = 0; j < 2; ++j) {
      const int n = nbase + j * 16 + lanelo;
#pragma unroll
      for (int r = 0; r < 8; ++r) {
        const int m = mbase + i * 16 + hi * 8 + r;
        if (m < 784) {
          const int oh = m / 28, ow = m % 28;
          const unsigned char mk = masks[b * 196 + (oh >> 1) * 14 + (ow >> 1)];
          const float v = mk ? fmaxf(acc[i][j][r], 0.f) : 0.f;
          X2[(((size_t)b * 768 + n) * 28 + oh) * 28 + ow] = f32_to_bf16(v);
        }
      }
    }
  }
}

// ---------------------------------------------------------------------------
// Kernel 4: MaxBlurPool 28 -> 14 fused with NCHW -> N(HW)C transpose, f32 out.
// out: [64][196][768] f32.
// ---------------------------------------------------------------------------
__global__ void maxblurpool2_out(const unsigned short* __restrict__ X2,
                                 float* __restrict__ out) {
  int i = blockIdx.x * blockDim.x + threadIdx.x;
  if (i >= 64 * 768 * 196) return;
  const int p = i % 196;
  const int n = (i / 196) % 768;
  const int b = i / (196 * 768);
  const int h = p / 14, w = p % 14;
  const unsigned short* src = X2 + ((size_t)b * 768 + n) * 784;
  const float kb[3] = {0.25f, 0.5f, 0.25f};
  float acc = 0.f;
#pragma unroll
  for (int t = 0; t < 3; ++t) {
    const int r = 2 * h - 1 + t;
    if ((unsigned)r > 26u) continue;
#pragma unroll
    for (int u = 0; u < 3; ++u) {
      const int s = 2 * w - 1 + u;
      if ((unsigned)s > 26u) continue;
      float mx = fmaxf(
          fmaxf(bf16_to_f32(src[r * 28 + s]), bf16_to_f32(src[r * 28 + s + 1])),
          fmaxf(bf16_to_f32(src[(r + 1) * 28 + s]),
                bf16_to_f32(src[(r + 1) * 28 + s + 1])));
      acc += kb[t] * kb[u] * mx;
    }
  }
  out[((size_t)b * 196 + p) * 768 + n] = acc;
}

// ---------------------------------------------------------------------------
extern "C" void kernel_launch(void* const* d_in, const int* in_sizes, int n_in,
                              void* d_out, int out_size, void* d_ws,
                              size_t ws_size, hipStream_t stream) {
  (void)in_sizes; (void)n_in; (void)out_size; (void)ws_size;

  const float*         x     = (const float*)d_in[0];
  const unsigned char* masks = (const unsigned char*)d_in[1];  // bool -> 1 byte
  const float*         w1    = (const float*)d_in[2];
  const float*         w2    = (const float*)d_in[3];
  float*               out   = (float*)d_out;

  // Workspace carve (256B aligned): ~211 MB total.
  char* ws = (char*)d_ws;
  size_t off = 0;
  auto carve = [&](size_t bytes) {
    void* p = ws + off;
    off += (bytes + 255) & ~(size_t)255;
    return p;
  };
  unsigned short* w2b = (unsigned short*)carve((size_t)768 * 3136 * 2);
  unsigned short* X1  = (unsigned short*)carve((size_t)64 * 64 * 112 * 112 * 2);
  unsigned short* X1p = (unsigned short*)carve((size_t)64 * 64 * 56 * 56 * 2);
  unsigned short* X2  = (unsigned short*)carve((size_t)64 * 768 * 28 * 28 * 2);

  {
    int n = 768 * 3136;
    cvt_w2<<<(n + 255) / 256, 256, 0, stream>>>(w2, w2b, n);
  }
  {
    dim3 g(7, 7, 64 * 64), blk(16, 16);
    conv1_mask_relu<<<g, blk, 0, stream>>>(x, masks, w1, X1);
  }
  {
    int n = 64 * 64 * 56 * 56;
    maxblurpool1<<<(n + 255) / 256, 256, 0, stream>>>(X1, X1p);
  }
  {
    dim3 g((784 + TILE_M - 1) / TILE_M, 768 / TILE_N, 64);
    conv2_wmma<<<g, 256, 0, stream>>>(X1p, w2b, masks, X2);
  }
  {
    int n = 64 * 768 * 196;
    maxblurpool2_out<<<(n + 255) / 256, 256, 0, stream>>>(X2, out);
  }
}
